// ATTNLM_82875688944241
// MI455X (gfx1250) — matrix operand order repaired and checked by reading, not compiled, not run on hardware
//
#include <hip/hip_runtime.h>
#include <math.h>
#include <stdint.h>

typedef __attribute__((ext_vector_type(16))) _Float16 v16h;
typedef __attribute__((ext_vector_type(8)))  _Float16 v8h;
typedef __attribute__((ext_vector_type(8)))  float    v8f;
typedef __attribute__((ext_vector_type(4)))  float    v4f;
typedef __attribute__((ext_vector_type(2)))  float    v2f;
typedef __attribute__((ext_vector_type(4)))  unsigned int u32x4;
typedef __attribute__((ext_vector_type(8)))  int      i32x8;
typedef __attribute__((ext_vector_type(4)))  int      i32x4;

#define NB    16
#define LSEQ  1024
#define DH    512
#define NCLS  64
#define APAD  520   // 512 + 8 f16 pad -> row stride 1040B (16B aligned, banks spread)

// ---- CDNA5 async / TDM feature probes -----------------------------------
#if __has_builtin(__builtin_amdgcn_global_load_async_to_lds_b128)
#define HAVE_ASYNC 1
// builtin signature (from hipcc diagnostic): params are pointers to
// int __attribute__((vector_size(16))) in AS(1) (global) and AS(3) (LDS).
typedef __attribute__((address_space(1))) i32x4* g4_t;
typedef __attribute__((address_space(3))) i32x4* l4_t;
// flat->AS casts via integers: global flat == 64-bit addr; LDS offset == low 32 bits
#define ASYNC_CP16(g, l)                                                     \
    __builtin_amdgcn_global_load_async_to_lds_b128(                          \
        (g4_t)(uintptr_t)(g), (l4_t)(unsigned)(uintptr_t)(l), 0, 0)
#endif

#if __has_builtin(__builtin_amdgcn_s_wait_asynccnt)
#define WAIT_ASYNC0() __builtin_amdgcn_s_wait_asynccnt(0)
#else
#define WAIT_ASYNC0() asm volatile("s_wait_asynccnt 0x0" ::: "memory")
#endif

#if __has_builtin(__builtin_amdgcn_tensor_load_to_lds) && \
    __has_builtin(__builtin_amdgcn_s_wait_tensorcnt)
#define HAVE_TDM 1
#endif

// ---- fragment helpers ---------------------------------------------------
// A fragment (16x32 f16, row = lane&15): half h holds K {8h..8h+7, 16+8h..23+8h}
__device__ __forceinline__ v16h frag_a(const _Float16* rowp, int ks, int h) {
    const int k0 = ks * 32 + h * 8;
    v8h lo = *(const v8h*)(rowp + k0);
    v8h hi = *(const v8h*)(rowp + k0 + 16);
    return __builtin_shufflevector(lo, hi, 0,1,2,3,4,5,6,7,8,9,10,11,12,13,14,15);
}
// B fragment (32x16 f16, col = lane&15, B[k][n] = W[n][k]): half h holds K {8h..8h+7, 16+8h..23+8h}
__device__ __forceinline__ v16h frag_b(const _Float16* wrow, int ks, int h) {
    const int k0 = ks * 32 + h * 8;
    v8h lo = *(const v8h*)(wrow + k0);
    v8h hi = *(const v8h*)(wrow + k0 + 16);
    return __builtin_shufflevector(lo, hi, 0,1,2,3,4,5,6,7,8,9,10,11,12,13,14,15);
}

// ---- weight prep: f32 -> f16, bias sum ----------------------------------
__global__ void k_prep(const float* __restrict__ Wih, const float* __restrict__ Whh,
                       const float* __restrict__ bih, const float* __restrict__ bhh,
                       _Float16* __restrict__ Wi16, _Float16* __restrict__ Wh16,
                       float* __restrict__ bsum, int nw, int nb) {
    int i = blockIdx.x * blockDim.x + threadIdx.x;
    if (i < nw) { Wi16[i] = (_Float16)Wih[i]; Wh16[i] = (_Float16)Whh[i]; }
    if (i < nb) { bsum[i] = bih[i] + bhh[i]; }
}

// ---- input projection: XS[m][n] = A[m] @ W^T + b  (M=16384, N=K=512) ----
// grid (1024, 4), block 256 (8 waves).
// Layer 0: Af32 = emb with token gather (f32->f16 convert while staging).
// Layer 1: Af16 = OUT16 (already f16); staged via TDM tensor_load_to_lds.
__global__ __launch_bounds__(256) void k_proj(const float* __restrict__ Af32,
                                              const _Float16* __restrict__ Af16,
                                              const int* __restrict__ tokens,
                                              const _Float16* __restrict__ W16,
                                              const float* __restrict__ bias,
                                              float* __restrict__ XS) {
    __shared__ _Float16 A16[16][APAD];
    const int tid = threadIdx.x;
    const int m0  = blockIdx.x * 16;

    if (Af16) {
#if defined(HAVE_TDM)
        if ((tid >> 5) == 0) {   // wave 0 issues one TDM descriptor for the tile
            const uint64_t ga = (uint64_t)(uintptr_t)(Af16 + (size_t)m0 * DH);
            const unsigned la = (unsigned)(uintptr_t)&A16[0][0];
            // D# group0: count=1 | lds_addr | global_addr(57b) | type=2
            u32x4 g0 = { 1u, la, (unsigned)(ga & 0xFFFFFFFFu),
                         ((unsigned)((ga >> 32) & 0x1FFFFFFu)) | (2u << 30) };
            // D# group1: data_size=2B, pad_enable, pad_interval=256dw, pad_amount=4dw
            // tensor_dim0=512, tensor_dim1=16, tile_dim0=512, tile_dim1=16,
            // tensor_dim0_stride=512
            i32x8 g1 = { 0x07D10000, (int)(512u << 16), (int)(16u << 16),
                         (int)(512u << 16), 16, 512, 0, 0 };
            i32x4 z4 = { 0, 0, 0, 0 };
#if __clang_major__ >= 23
            i32x8 z8 = { 0, 0, 0, 0, 0, 0, 0, 0 };
            __builtin_amdgcn_tensor_load_to_lds(g0, g1, z4, z4, z8, 0);
#else
            __builtin_amdgcn_tensor_load_to_lds(g0, g1, z4, z4, 0);
#endif
            __builtin_amdgcn_s_wait_tensorcnt(0);
        }
#else
        const int row = tid >> 4, seg = tid & 15;
        const _Float16* s = Af16 + (size_t)(m0 + row) * DH + seg * 32;
        #pragma unroll
        for (int j = 0; j < 4; ++j)
            *(v8h*)&A16[row][seg * 32 + j * 8] = *(const v8h*)(s + j * 8);
#endif
    } else {   // gathered f32 rows, convert to f16 while staging
        const int row = tid >> 4, seg = tid & 15, m = m0 + row;
        const float* src = Af32 + (size_t)tokens[m] * DH;
        const int c0 = seg * 32;
        #pragma unroll
        for (int j = 0; j < 8; ++j) {
            v4f v = *(const v4f*)(src + c0 + j * 4);
            A16[row][c0 + j*4 + 0] = (_Float16)v.x;
            A16[row][c0 + j*4 + 1] = (_Float16)v.y;
            A16[row][c0 + j*4 + 2] = (_Float16)v.z;
            A16[row][c0 + j*4 + 3] = (_Float16)v.w;
        }
    }
    __syncthreads();

    const int lane = tid & 31, wv = tid >> 5;
    const int h = lane >> 4, r = lane & 15;
    const int n0 = (blockIdx.y * 8 + wv) * 16;
    const _Float16* wrow = W16 + (size_t)(n0 + r) * DH;
    const _Float16* arow = &A16[r][0];

    v8f acc = {};
    #pragma unroll
    for (int ks = 0; ks < 16; ++ks) {
        v16h a = frag_a(arow, ks, h);
        v16h b = frag_b(wrow, ks, h);
        acc = __builtin_amdgcn_wmma_f32_16x16x32_f16(false, a, false, b,
                                                     (short)0, acc, false, false);
    }
    const int col = n0 + r;
    const float bv = bias[col];
    #pragma unroll
    for (int v = 0; v < 8; ++v) {
        const int row = v + h * 8;
        XS[(size_t)(m0 + row) * DH + col] = acc[v] + bv;
    }
}

// ---- sequential recurrence: H = tanh(XS[:,t,:] + H @ Wh^T), 1024 steps --
// One workgroup, 32 waves; wave w owns cols [16w,16w+16); Wh^T tile persistent
// in VGPRs (16 x v16h); H lives in LDS (f16). Next step's XS slice is
// prefetched into LDS with async global->LDS copies, hidden under the WMMAs.
__global__ __launch_bounds__(1024) void k_rnn(const float* __restrict__ XS,
                                              const _Float16* __restrict__ Wh16,
                                              _Float16* __restrict__ OUT16,
                                              float* __restrict__ OUTf) {
    __shared__ _Float16 Hb[16][APAD];
#if defined(HAVE_ASYNC)
    __shared__ float xsl[16][DH];   // 32 KB staging for XS[:,t,:]
#endif
    const int tid = threadIdx.x;
    const int lane = tid & 31, wv = tid >> 5;
    const int h = lane >> 4, r = lane & 15;
    const int col = wv * 16 + r;

    // persistent B fragments: Wh^T 512x16 tile for this wave (128 VGPRs/lane)
    v16h bf[16];
    const _Float16* wrow = Wh16 + (size_t)col * DH;
    #pragma unroll
    for (int ks = 0; ks < 16; ++ks) bf[ks] = frag_b(wrow, ks, h);

    for (int i = tid; i < 16 * APAD; i += 1024) (&Hb[0][0])[i] = (_Float16)0.f;

#if defined(HAVE_ASYNC)
    // each thread owns two 16B chunks of the 32KB slice; flat offset == LDS offset
    const int f0 = tid * 16, f1 = (tid + 1024) * 16;
    const char* gp0 = (const char*)XS + (size_t)(f0 >> 11) * LSEQ * 2048 + (f0 & 2047);
    const char* gp1 = (const char*)XS + (size_t)(f1 >> 11) * LSEQ * 2048 + (f1 & 2047);
    char* lp0 = (char*)&xsl[0][0] + f0;
    char* lp1 = (char*)&xsl[0][0] + f1;
    ASYNC_CP16(gp0, lp0);            // prime slice t=0
    ASYNC_CP16(gp1, lp1);
#endif
    __syncthreads();

    for (int t = 0; t < LSEQ; ++t) {
        const _Float16* arow = &Hb[r][0];
        v8f acc = {};
        #pragma unroll
        for (int ks = 0; ks < 16; ++ks)
            acc = __builtin_amdgcn_wmma_f32_16x16x32_f16(false, frag_a(arow, ks, h),
                                                         false, bf[ks],
                                                         (short)0, acc, false, false);
#if defined(HAVE_ASYNC)
        WAIT_ASYNC0();               // slice t arrived (issued one step earlier)
#endif
        __syncthreads();             // A: slice visible; all Hb reads done
        #pragma unroll
        for (int v = 0; v < 8; ++v) {
            const int row = v + h * 8;                     // batch index
            const size_t idx = ((size_t)row * LSEQ + t) * DH + col;
#if defined(HAVE_ASYNC)
            float val = tanhf(acc[v] + xsl[row][col]);
#else
            float val = tanhf(acc[v] + XS[idx]);
            __builtin_prefetch(&XS[idx + DH], 0, 1);
#endif
            if (OUTf)  OUTf[idx]  = val;
            if (OUT16) OUT16[idx] = (_Float16)val;
            Hb[row][col] = (_Float16)val;
        }
        __syncthreads();             // B: Hb writes visible; slice consumed
#if defined(HAVE_ASYNC)
        if (t + 1 < LSEQ) {          // prefetch slice t+1 under next step's WMMAs
            gp0 += 2048; gp1 += 2048;
            ASYNC_CP16(gp0, lp0);
            ASYNC_CP16(gp1, lp1);
        }
#endif
    }
}

// ---- single attention row per batch (at t = eos[b]) + softmax + mix -----
__global__ __launch_bounds__(256) void k_attn(const float* __restrict__ OUT,
                                              const int* __restrict__ eos,
                                              float* __restrict__ attz,
                                              float* __restrict__ zb) {
    __shared__ float q[DH];
    __shared__ float sc[LSEQ];
    __shared__ float red[256];
    const int b = blockIdx.x, tid = threadIdx.x;
    const int t = eos[b];
    const float* Ob = OUT + (size_t)b * LSEQ * DH;

    for (int k = tid; k < DH; k += 256) q[k] = Ob[(size_t)t * DH + k];
    __syncthreads();

    for (int s = tid; s < LSEQ; s += 256) {
        const float* row = Ob + (size_t)s * DH;
        float d = 0.f;
        for (int k = 0; k < DH; k += 4) {
            v4f a = *(const v4f*)(row + k);
            d += a.x * q[k] + a.y * q[k+1] + a.z * q[k+2] + a.w * q[k+3];
        }
        sc[s] = (s < t) ? d : -1.0e9f;   // strict-causal mask incl. diagonal
    }
    __syncthreads();

    float m = -3.4e38f;
    for (int s = tid; s < LSEQ; s += 256) m = fmaxf(m, sc[s]);
    red[tid] = m; __syncthreads();
    for (int st = 128; st > 0; st >>= 1) {
        if (tid < st) red[tid] = fmaxf(red[tid], red[tid + st]);
        __syncthreads();
    }
    const float mx = red[0];
    __syncthreads();

    float ssum = 0.f;
    for (int s = tid; s < LSEQ; s += 256) { float e = __expf(sc[s] - mx); sc[s] = e; ssum += e; }
    red[tid] = ssum; __syncthreads();
    for (int st = 128; st > 0; st >>= 1) {
        if (tid < st) red[tid] += red[tid + st];
        __syncthreads();
    }
    const float inv = 1.0f / red[0];
    __syncthreads();

    for (int k = tid; k < DH; k += 256) {
        float a0 = 0.f;
        for (int s = 0; s < LSEQ; ++s) a0 += sc[s] * Ob[(size_t)s * DH + k];
        attz[b * DH + k] = a0 * inv;
        zb[b * DH + k]   = q[k];
    }
}

// ---- head: dec = [attz|z] @ Wc^T + bc ; logit = dec @ Wd^T + bd ---------
// Full-precision V_WMMA_F32_16X16X4_F32 (M=16 == batch).
__global__ __launch_bounds__(1024) void k_head(const float* __restrict__ attz,
                                               const float* __restrict__ zb,
                                               const float* __restrict__ Wc,
                                               const float* __restrict__ bc,
                                               const float* __restrict__ Wd,
                                               const float* __restrict__ bd,
                                               float* __restrict__ out) {
    __shared__ float dec[16][DH + 8];
    const int tid = threadIdx.x;
    const int lane = tid & 31, wv = tid >> 5;
    const int h = lane >> 4, r = lane & 15;
    {   // phase 1: 32 waves, 32 n-tiles of dec (K = 1024)
        const int n0 = wv * 16;
        const float* wcrow = Wc + (size_t)(n0 + r) * (2 * DH);
        v8f acc = {};
        for (int kk = 0; kk < DH; kk += 4) {   // k in [0,512): att_z half
            v2f a = *(const v2f*)(attz + r * DH + kk + 2 * h);
            v2f b = *(const v2f*)(wcrow + kk + 2 * h);
            acc = __builtin_amdgcn_wmma_f32_16x16x4_f32(false, a, false, b,
                                                        (short)0, acc, false, false);
        }
        for (int kk = 0; kk < DH; kk += 4) {   // k in [512,1024): z half
            v2f a = *(const v2f*)(zb + r * DH + kk + 2 * h);
            v2f b = *(const v2f*)(wcrow + DH + kk + 2 * h);
            acc = __builtin_amdgcn_wmma_f32_16x16x4_f32(false, a, false, b,
                                                        (short)0, acc, false, false);
        }
        const float bv = bc[n0 + r];
        #pragma unroll
        for (int v = 0; v < 8; ++v) dec[v + 8 * h][n0 + r] = acc[v] + bv;
    }
    __syncthreads();
    if (wv < 4) {   // phase 2: 4 n-tiles of logit (K = 512)
        const int n0 = wv * 16;
        const float* wdrow = Wd + (size_t)(n0 + r) * DH;
        v8f acc = {};
        for (int kk = 0; kk < DH; kk += 4) {
            v2f a = *(const v2f*)(&dec[r][kk + 2 * h]);
            v2f b = *(const v2f*)(wdrow + kk + 2 * h);
            acc = __builtin_amdgcn_wmma_f32_16x16x4_f32(false, a, false, b,
                                                        (short)0, acc, false, false);
        }
        const float bv = bd[n0 + r];
        #pragma unroll
        for (int v = 0; v < 8; ++v) out[(v + 8 * h) * NCLS + n0 + r] = acc[v] + bv;
    }
}

// ---- host-side orchestration --------------------------------------------
extern "C" void kernel_launch(void* const* d_in, const int* in_sizes, int n_in,
                              void* d_out, int out_size, void* d_ws, size_t ws_size,
                              hipStream_t stream) {
    (void)in_sizes; (void)n_in; (void)out_size; (void)ws_size;
    const int*   x   = (const int*)d_in[0];
    const int*   eos = (const int*)d_in[1];
    const float* emb = (const float*)d_in[2];
    const float* Wih = (const float*)d_in[3];
    const float* Whh = (const float*)d_in[4];
    const float* bih = (const float*)d_in[5];
    const float* bhh = (const float*)d_in[6];
    const float* Wc  = (const float*)d_in[7];
    const float* bc  = (const float*)d_in[8];
    const float* Wd  = (const float*)d_in[9];
    const float* bd  = (const float*)d_in[10];
    float* out = (float*)d_out;

    char* ws = (char*)d_ws;
    const size_t NW = (size_t)2 * DH * DH;          // per-weight-array elems
    _Float16* Wi16 = (_Float16*)ws;  ws += NW * sizeof(_Float16);
    _Float16* Wh16 = (_Float16*)ws;  ws += NW * sizeof(_Float16);
    float* bsum = (float*)ws;        ws += 2 * DH * sizeof(float);
    float* attz = (float*)ws;        ws += NB * DH * sizeof(float);
    float* zbuf = (float*)ws;        ws += NB * DH * sizeof(float);
    _Float16* out16 = (_Float16*)ws; ws += (size_t)NB * LSEQ * DH * sizeof(_Float16);
    float* XS   = (float*)ws;        ws += (size_t)NB * LSEQ * DH * sizeof(float);
    float* OUTb = (float*)ws;        // layer-1 output (f32, read by attention)

    k_prep<<<(int)((NW + 255) / 256), 256, 0, stream>>>(Wih, Whh, bih, bhh,
                                                        Wi16, Wh16, bsum,
                                                        (int)NW, 2 * DH);
    const dim3 gp(LSEQ * NB / 16, 4);
    // layer 0: gather + proj, then scan (emit f16-only output for layer-1 GEMM)
    k_proj<<<gp, 256, 0, stream>>>(emb, nullptr, x, Wi16, bsum, XS);
    k_rnn<<<1, 1024, 0, stream>>>(XS, Wh16, out16, nullptr);
    // layer 1: proj from f16 layer-0 output (TDM-staged A), then scan
    k_proj<<<gp, 256, 0, stream>>>(nullptr, out16, nullptr, Wi16 + DH * DH,
                                   bsum + DH, XS);
    k_rnn<<<1, 1024, 0, stream>>>(XS, Wh16 + DH * DH, nullptr, OUTb);
    // eos-row attention + head
    k_attn<<<NB, 256, 0, stream>>>(OUTb, eos, attz, zbuf);
    k_head<<<1, 1024, 0, stream>>>(attz, zbuf, Wc, bc, Wd, bd, out);
}
